// FactorizedVQBottleneck_84284438217387
// MI455X (gfx1250) — compile-verified, matrix operand order checked by reading, not run on hardware
//
#include <hip/hip_runtime.h>

#define CB_K   8192
#define CB_D   128
#define N_CB   2
#define N_T    1024
#define NROWS  8192            // B*T
#define Q_ELEMS 2097152        // 8*256*1024
#define IDX_ELEMS 16384        // 8*2*1024
#define COMMIT_LAMBDA 0.25f

#define NSEG    8              // K-range segments (grid parallelism)
#define KSEG    (CB_K / NSEG)  // 1024 codes per segment
#define CHUNK   64             // codes staged in LDS per double-buffer slot
#define NCHUNK  (KSEG / CHUNK) // 16

typedef __attribute__((ext_vector_type(16))) __bf16         v16bf;
typedef __attribute__((ext_vector_type(16))) unsigned short v16us;
typedef __attribute__((ext_vector_type(8)))  unsigned short v8us;
typedef __attribute__((ext_vector_type(8)))  float          v8f;
typedef __attribute__((ext_vector_type(4)))  unsigned int   u32x4v;
typedef __attribute__((ext_vector_type(8)))  int            i32x8v;
typedef __attribute__((ext_vector_type(4)))  int            i32x4v;

__device__ __forceinline__ unsigned short f32_to_bf16_rn(float f) {
    unsigned u = __float_as_uint(f);
    u += 0x7FFFu + ((u >> 16) & 1u);          // round-to-nearest-even
    return (unsigned short)(u >> 16);
}
__device__ __forceinline__ float bf16_to_f32(unsigned short h) {
    return __uint_as_float(((unsigned)h) << 16);
}

// Tensor Data Mover: async copy of a 2D tile (CHUNK rows x 128 bf16, contiguous)
// from global into LDS.  D# built per CDNA5 ISA.
__device__ __forceinline__ void tdm_load_tile(const unsigned short* gsrc,
                                              unsigned lds_byte_off) {
    unsigned long long ga = (unsigned long long)(uintptr_t)gsrc;
    u32x4v g0 = { 1u,                                    // count=1 (valid)
                  lds_byte_off,                          // lds_addr
                  (unsigned)ga,                          // global_addr[31:0]
                  (unsigned)((ga >> 32) & 0x01FFFFFFu)   // global_addr[56:32]
                      | (2u << 30) };                    // type=2 ("image")
    i32x8v g1 = { (int)(1u << 16),              // data_size=1 -> 2-byte elems
                  (int)(128u << 16),            // tensor_dim0 = 128 (code row)
                  (int)((unsigned)CHUNK << 16), // tensor_dim1 = CHUNK rows
                  (int)(128u << 16),            // tile_dim0 = 128
                  (int)CHUNK,                   // tile_dim1 = CHUNK, dim2 = 0
                  128,                          // tensor_dim0_stride (contig)
                  0, 0 };
    i32x4v z4 = { 0, 0, 0, 0 };
#if __clang_major__ >= 23
    i32x8v z8 = { 0, 0, 0, 0, 0, 0, 0, 0 };
    __builtin_amdgcn_tensor_load_to_lds(g0, g1, z4, z4, z8, 0);
#else
    __builtin_amdgcn_tensor_load_to_lds(g0, g1, z4, z4, 0);
#endif
}

// ---------------------------------------------------------------------------
// Kernel 1: split codebooks fp32 -> bf16 hi/lo, ||c||^2, init argmin slots+loss.
// ---------------------------------------------------------------------------
__global__ __launch_bounds__(128)
void vq_prep(const float* __restrict__ cb,
             unsigned short* __restrict__ cbh,
             unsigned short* __restrict__ cbl,
             float* __restrict__ c2,
             unsigned long long* __restrict__ pk,
             float* __restrict__ loss_out) {
    __shared__ float red[128];
    const int row = blockIdx.x;               // n*K + k  (16384 blocks)
    const int d   = threadIdx.x;
    const size_t off = (size_t)row * CB_D + d;
    float v = cb[off];
    unsigned short h = f32_to_bf16_rn(v);
    cbh[off] = h;
    cbl[off] = f32_to_bf16_rn(v - bf16_to_f32(h));
    red[d] = v * v;
    __syncthreads();
    for (int s = 64; s > 0; s >>= 1) {
        if (d < s) red[d] += red[d + s];
        __syncthreads();
    }
    if (d == 0) {
        c2[row] = red[0];
        pk[row] = 0xFFFFFFFFFFFFFFFFULL;      // one slot per (n, bt) row
        if (row == 0) *loss_out = 0.0f;
    }
}

// ---------------------------------------------------------------------------
// Kernel 2: argmin over a K-segment. bf16x3-split WMMA; each wave owns TWO
// 16-row strips sharing every B fragment (halves LDS reads per WMMA while
// keeping two independent accumulator chains). TDM double-buffers B tiles.
// Segments merge via packed u64 atomicMin (order-preserving float key | idx).
// grid = (32 row-tiles of 256, NCB, NSEG), 256 threads = 8 waves.
// ---------------------------------------------------------------------------
__global__ __launch_bounds__(256)
void vq_argmin(const float* __restrict__ x,
               const unsigned short* __restrict__ cbh,
               const unsigned short* __restrict__ cbl,
               const float* __restrict__ c2,
               unsigned long long* __restrict__ pk) {
    // [buffer][hi/lo][CHUNK rows * 128]  = 64 KB total
    __shared__ __align__(16) unsigned short sB[2][2][CHUNK * CB_D];

    const int tile = blockIdx.x;              // 0..31 (256 rows each)
    const int n    = blockIdx.y;              // codebook
    const int seg  = blockIdx.z;              // K segment
    const int tid  = threadIdx.x;
    const int wave = tid >> 5;
    const int lane = tid & 31;
    const int l15  = lane & 15;
    const int hi16 = (lane >> 4) & 1;

    const int waveRow = tile * 256 + wave * 32;   // 2 strips of 16 rows
    // ---- register-resident A fragments: 2 strips x (hi,lo) x 4 K-chunks ----
    // 16-bit A 16x32 layout: lanes 0-15 -> M=lane, K {0..7,16..23};
    //                        lanes 16-31 -> M=lane-16, K {8..15,24..31}.
    v16bf ah[2][4], al[2][4];
    for (int s = 0; s < 2; ++s) {
        const int myrow = waveRow + s * 16 + l15;
        const int b = myrow >> 10;            // 256-row tile never straddles b
        const int t = myrow & 1023;
        for (int dk = 0; dk < 4; ++dk) {
            v16us th, tl;
            for (int e = 0; e < 16; ++e) {
                int dd = dk * 32 + hi16 * 8 + ((e < 8) ? e : (e + 8));
                float v = x[((size_t)(b * 256 + n * 128 + dd)) * 1024 + t];
                unsigned short h = f32_to_bf16_rn(v);
                th[e] = h;
                tl[e] = f32_to_bf16_rn(v - bf16_to_f32(h));
            }
            ah[s][dk] = __builtin_bit_cast(v16bf, th);
            al[s][dk] = __builtin_bit_cast(v16bf, tl);
        }
    }

    float minv[2][8];
    int   mini[2][8];
    for (int s = 0; s < 2; ++s)
        for (int r = 0; r < 8; ++r) { minv[s][r] = 3.4e38f; mini[s][r] = 0; }

    const int    segBase = seg * KSEG;
    const size_t cbBase  = ((size_t)n * CB_K + segBase) * CB_D;

    // prologue: stage chunk 0 into buffer 0 (TDM, TENSORcnt-tracked)
    if (wave == 0) {
        tdm_load_tile(cbh + cbBase, (unsigned)(uintptr_t)&sB[0][0][0]);
        tdm_load_tile(cbl + cbBase, (unsigned)(uintptr_t)&sB[0][1][0]);
    }

    for (int chunk = 0; chunk < NCHUNK; ++chunk) {
        const int buf = chunk & 1;
        __syncthreads();   // all waves done with buffer buf^1 -> safe to refill
        if (wave == 0) {
            if (chunk + 1 < NCHUNK) {
                const size_t nxt = cbBase + (size_t)(chunk + 1) * CHUNK * CB_D;
                tdm_load_tile(cbh + nxt, (unsigned)(uintptr_t)&sB[buf ^ 1][0][0]);
                tdm_load_tile(cbl + nxt, (unsigned)(uintptr_t)&sB[buf ^ 1][1][0]);
                __builtin_amdgcn_s_wait_tensorcnt(2);  // in-order: chunk landed
            } else {
                __builtin_amdgcn_s_wait_tensorcnt(0);
            }
        }
        __syncthreads();   // chunk's tile visible to all waves

        const unsigned short* sH = &sB[buf][0][0];
        const unsigned short* sL = &sB[buf][1][0];

        for (int kt = 0; kt < CHUNK / 16; ++kt) {      // 16-code column tiles
            v8f acc0 = {0.f,0.f,0.f,0.f,0.f,0.f,0.f,0.f};
            v8f acc1 = {0.f,0.f,0.f,0.f,0.f,0.f,0.f,0.f};
            const int rb = (kt * 16 + l15) * CB_D;     // B column = code row
            for (int dk = 0; dk < 4; ++dk) {
                const int o = rb + dk * 32 + hi16 * 8;
                v16bf bh = __builtin_bit_cast(v16bf, __builtin_shufflevector(
                    *(const v8us*)&sH[o], *(const v8us*)&sH[o + 16],
                    0,1,2,3,4,5,6,7,8,9,10,11,12,13,14,15));
                v16bf bl = __builtin_bit_cast(v16bf, __builtin_shufflevector(
                    *(const v8us*)&sL[o], *(const v8us*)&sL[o + 16],
                    0,1,2,3,4,5,6,7,8,9,10,11,12,13,14,15));
                // S = Xh*Ch + Xh*Cl + Xl*Ch ; both strips share bh/bl
                acc0 = __builtin_amdgcn_wmma_f32_16x16x32_bf16(
                           false, ah[0][dk], false, bh, (short)0, acc0, false, false);
                acc1 = __builtin_amdgcn_wmma_f32_16x16x32_bf16(
                           false, ah[1][dk], false, bh, (short)0, acc1, false, false);
                acc0 = __builtin_amdgcn_wmma_f32_16x16x32_bf16(
                           false, ah[0][dk], false, bl, (short)0, acc0, false, false);
                acc1 = __builtin_amdgcn_wmma_f32_16x16x32_bf16(
                           false, ah[1][dk], false, bl, (short)0, acc1, false, false);
                acc0 = __builtin_amdgcn_wmma_f32_16x16x32_bf16(
                           false, al[0][dk], false, bh, (short)0, acc0, false, false);
                acc1 = __builtin_amdgcn_wmma_f32_16x16x32_bf16(
                           false, al[1][dk], false, bh, (short)0, acc1, false, false);
            }
            const int   code = segBase + chunk * CHUNK + kt * 16 + l15;
            const float c2v  = c2[n * CB_K + code];    // shared by both strips
            for (int r = 0; r < 8; ++r) {
                float m0 = __builtin_fmaf(-2.0f, acc0[r], c2v);
                float m1 = __builtin_fmaf(-2.0f, acc1[r], c2v);
                if (m0 < minv[0][r]) { minv[0][r] = m0; mini[0][r] = code; }
                if (m1 < minv[1][r]) { minv[1][r] = m1; mini[1][r] = code; }
            }
        }
    }

    // min across the 16 lanes of each half-group, then merge across segments.
    for (int s = 0; s < 2; ++s) {
        for (int r = 0; r < 8; ++r) {
            float v = minv[s][r];
            int   i = mini[s][r];
            for (int m = 8; m >= 1; m >>= 1) {
                float ov = __shfl_xor(v, m, 32);
                int   oi = __shfl_xor(i, m, 32);
                if (ov < v || (ov == v && oi < i)) { v = ov; i = oi; }
            }
            if (l15 == 0) {
                int row = waveRow + s * 16 + r + 8 * hi16;  // VGPR r -> M
                unsigned vb  = __float_as_uint(v);
                unsigned key = (vb & 0x80000000u) ? ~vb : (vb | 0x80000000u);
                unsigned long long packed =
                    ((unsigned long long)key << 32) | (unsigned)i;
                atomicMin(&pk[n * NROWS + row], packed);
            }
        }
    }
}

// ---------------------------------------------------------------------------
// Kernel 3: gather fp32 codes -> quantized (B,C,T), indices (float), loss.
// ---------------------------------------------------------------------------
__global__ __launch_bounds__(256)
void vq_gather(const float* __restrict__ x,
               const float* __restrict__ cb,
               const unsigned long long* __restrict__ pk,
               float* __restrict__ outq,
               float* __restrict__ outidx,
               float* __restrict__ loss) {
    const int g  = blockIdx.x * 256 + threadIdx.x;   // 0..16383
    const int n  = g >> 13;
    const int bt = g & (NROWS - 1);
    const int b  = bt >> 10;
    const int t  = bt & 1023;
    const int idx = (int)(pk[n * NROWS + bt] & 0xFFFFFFFFu);
    const float* crow = cb + ((size_t)n * CB_K + idx) * CB_D;
    float sum = 0.f;
    for (int dd = 0; dd < CB_D; ++dd) {
        const size_t a = ((size_t)(b * 256 + n * 128 + dd)) * 1024 + t;
        float q  = crow[dd];
        float xv = x[a];
        outq[a] = q;                          // straight-through forward == q
        float dql = xv - q;
        sum = __builtin_fmaf(dql, dql, sum);
    }
    outidx[(size_t)b * (N_CB * N_T) + n * N_T + t] = (float)idx;
    for (int m = 16; m >= 1; m >>= 1) sum += __shfl_xor(sum, m, 32);
    if ((threadIdx.x & 31) == 0)
        unsafeAtomicAdd(loss, sum * (COMMIT_LAMBDA / (float)(NROWS * CB_D)));
}

// ---------------------------------------------------------------------------
extern "C" void kernel_launch(void* const* d_in, const int* in_sizes, int n_in,
                              void* d_out, int out_size, void* d_ws, size_t ws_size,
                              hipStream_t stream) {
    const float* x  = (const float*)d_in[0];   // (8,256,1024) f32
    const float* cb = (const float*)d_in[1];   // (2,8192,128) f32

    float* out     = (float*)d_out;
    float* outq    = out;                       // 2,097,152
    float* outidx  = out + Q_ELEMS;             // 16,384 (indices as float)
    float* outloss = out + Q_ELEMS + IDX_ELEMS; // 1

    // workspace: cbh | cbl | c2 | packed-argmin   (~8.3 MB)
    unsigned short* cbh = (unsigned short*)d_ws;
    unsigned short* cbl = cbh + (size_t)N_CB * CB_K * CB_D;
    float* c2 = (float*)(cbl + (size_t)N_CB * CB_K * CB_D);
    unsigned long long* pkb = (unsigned long long*)(c2 + N_CB * CB_K);

    vq_prep  <<<dim3(N_CB * CB_K),     dim3(128), 0, stream>>>(cb, cbh, cbl, c2, pkb, outloss);
    vq_argmin<<<dim3(32, N_CB, NSEG),  dim3(256), 0, stream>>>(x, cbh, cbl, c2, pkb);
    vq_gather<<<dim3(64),              dim3(256), 0, stream>>>(x, cb, pkb, outq, outidx, outloss);
}
